// GraphModel_48017734369581
// MI455X (gfx1250) — matrix operand order repaired and checked by reading, not compiled, not run on hardware
//
#include <hip/hip_runtime.h>
#include <stdint.h>

// ---------------- types ----------------
typedef __bf16 bf16;
typedef bf16  v16bf __attribute__((ext_vector_type(16)));
typedef float v8f   __attribute__((ext_vector_type(8)));

union FragB16 { v16bf v; uint4 q[2]; };

__device__ __forceinline__ unsigned short f2bfu(float f) {
    unsigned u = __builtin_bit_cast(unsigned, f);
    u += 0x7fffu + ((u >> 16) & 1u);          // round-to-nearest-even
    return (unsigned short)(u >> 16);
}
__device__ __forceinline__ unsigned pack2(float x, float y) {
    return (unsigned)f2bfu(x) | ((unsigned)f2bfu(y) << 16);
}
__device__ __forceinline__ void atomic_add_f32(float* p, float v) {
    __hip_atomic_fetch_add(p, v, __ATOMIC_RELAXED, __HIP_MEMORY_SCOPE_AGENT);
}

// ---------------- degree / norm ----------------
__global__ void k_deg_init(float* __restrict__ deg, int n) {
    int i = blockIdx.x * 256 + threadIdx.x;
    if (i < n) deg[i] = 1.0f;                 // self-loop
}
__global__ void k_deg_count(const long long* __restrict__ dst, float* __restrict__ deg, int E) {
    int e = blockIdx.x * 256 + threadIdx.x;
    if (e < E) atomic_add_f32(&deg[(int)dst[e]], 1.0f);
}
__global__ void k_rsqrt(float* __restrict__ deg, int n) {
    int i = blockIdx.x * 256 + threadIdx.x;
    if (i < n) deg[i] = rsqrtf(deg[i]);       // deg >= 1 always (self-loops)
}

// ---------------- W pre-pack into WMMA B-fragment order ----------------
// wf[((kc*8 + nt)*32 + lane)*16 + e] = bf16(W[(kc*32 + e + 16*(lane>>4)) * 128 + nt*16 + (lane&15)])
// B-fragment (32x16 bf16): lane holds col N = lane%16; this code packs elem e -> K = e + 16*(lane/16), same order as A.
__global__ void k_pack_w(const float* __restrict__ W, unsigned short* __restrict__ wf, int K) {
    int idx = blockIdx.x * 256 + threadIdx.x;
    if (idx >= K * 128) return;
    int e    = idx & 15;
    int lane = (idx >> 4) & 31;
    int nt   = (idx >> 9) & 7;
    int kc   = idx >> 12;
    int k = kc * 32 + e + ((lane >> 4) * 16);
    int n = nt * 16 + (lane & 15);
    wf[idx] = f2bfu(W[k * 128 + n]);
}

// ---------------- GEMM with WMMA bf16 ----------------
// t' = dinv[row] * (A[nrows x K] @ W[K x 128]); also writes acc init = t' (self loop).
// Block: 256 threads (8 waves). Block tile 64(M) x 128(N). Wave = 16(M) x 64(N).
#define A_STRIDE 40   // padded LDS row stride in halves: 80B (16B-aligned, conflict-free)

template <int K>
__global__ __launch_bounds__(256)
void gemm_wmma(const float* __restrict__ A, const unsigned short* __restrict__ wf,
               const float* __restrict__ dinv,
               float* __restrict__ tp, float* __restrict__ acc, int nrows)
{
    __shared__ __align__(16) unsigned short As[64 * A_STRIDE];

    const int tid  = threadIdx.x;
    const int lane = tid & 31;
    const int wave = tid >> 5;
    const int wm   = wave & 3;     // 0..3 : 16-row tile
    const int wn   = wave >> 2;    // 0..1 : 64-col half
    const int block_m = blockIdx.x * 64;

    v8f c[4] = {};                 // four 16x16 f32 accumulators

    for (int kc = 0; kc < K / 32; ++kc) {
        const int k0 = kc * 32;
        // ---- stage A tile: 64 rows x 32 k, 512 float4 loads, 2 per thread
        #pragma unroll
        for (int i = 0; i < 2; ++i) {
            int f    = tid + i * 256;       // 0..511
            int row  = f >> 3;              // 8 float4 per row
            int col4 = f & 7;
            int grow = block_m + row; if (grow >= nrows) grow = nrows - 1;
            const float4 v = ((const float4*)(A + (size_t)grow * K + k0))[col4];
            uint2 p; p.x = pack2(v.x, v.y); p.y = pack2(v.z, v.w);
            *(uint2*)&As[row * A_STRIDE + col4 * 4] = p;
        }
        __syncthreads();

        // ---- A fragment (16x32 bf16): lane holds row M = lane%16;
        // elem e -> K = e + 16*(lane/16)
        // => two contiguous 8-elem runs: 2x ds_load_b128
        FragB16 fa;
        {
            const int arow = wm * 16 + (lane & 15);
            const int koff = (lane >> 4) * 16;
            const unsigned short* ap = &As[arow * A_STRIDE + koff];
            fa.q[0] = *(const uint4*)(ap);        // K = koff..koff+7
            fa.q[1] = *(const uint4*)(ap + 8);    // K = koff+8..koff+15
        }
        // ---- B fragments: pre-packed, per-lane contiguous: 2x global_load_b128 (L2-hot)
        #pragma unroll
        for (int nt = 0; nt < 4; ++nt) {
            FragB16 fb;
            const unsigned short* bp =
                wf + ((((size_t)kc * 8 + wn * 4 + nt) * 32 + lane) * 16);
            fb.q[0] = ((const uint4*)bp)[0];
            fb.q[1] = ((const uint4*)bp)[1];
            c[nt] = __builtin_amdgcn_wmma_f32_16x16x32_bf16(
                        false, fa.v, false, fb.v, (short)0, c[nt], false, false);
        }
        __syncthreads();
    }

    // ---- epilogue: C/D layout VGPR v -> M = v + 8*(lane/16), N = lane%16
    const int hl   = lane >> 4;
    const int mrow = block_m + wm * 16 + hl * 8;   // first of this lane's 8 rows
    float dv[8];
    #pragma unroll
    for (int v = 0; v < 8; ++v) {                  // hoisted dinv loads (clamped)
        int r = mrow + v; if (r >= nrows) r = nrows - 1;
        dv[v] = dinv[r];
    }
    if (block_m + 64 <= nrows) {
        // full tile: uniform branch, no predication; lanes 0..15 write 16
        // consecutive cols -> coalesced 64B runs per store
        #pragma unroll
        for (int nt = 0; nt < 4; ++nt) {
            const int col = wn * 64 + nt * 16 + (lane & 15);
            #pragma unroll
            for (int v = 0; v < 8; ++v) {
                float val = c[nt][v] * dv[v];
                size_t idx = (size_t)(mrow + v) * 128 + col;
                tp[idx]  = val;     // message buffer t'
                acc[idx] = val;     // accumulator init (self-loop term)
            }
        }
    } else {
        #pragma unroll
        for (int nt = 0; nt < 4; ++nt) {
            const int col = wn * 64 + nt * 16 + (lane & 15);
            #pragma unroll
            for (int v = 0; v < 8; ++v) {
                if (mrow + v < nrows) {
                    float val = c[nt][v] * dv[v];
                    size_t idx = (size_t)(mrow + v) * 128 + col;
                    tp[idx]  = val;
                    acc[idx] = val;
                }
            }
        }
    }
}

// ---------------- edge scatter: one wave per edge ----------------
__global__ __launch_bounds__(256)
void k_scatter(const long long* __restrict__ src, const long long* __restrict__ dst,
               const float* __restrict__ tp, float* __restrict__ acc, int E)
{
    int e = blockIdx.x * 8 + (threadIdx.x >> 5);   // 8 edges per block
    if (e >= E) return;
    int lane = threadIdx.x & 31;
    int s = (int)src[e];
    int d = (int)dst[e];
    float4 v = ((const float4*)(tp + (size_t)s * 128))[lane];
    float* o = acc + (size_t)d * 128 + lane * 4;
    atomic_add_f32(o + 0, v.x);
    atomic_add_f32(o + 1, v.y);
    atomic_add_f32(o + 2, v.z);
    atomic_add_f32(o + 3, v.w);
}

// ---------------- finalize: h = [relu](dinv[i]*acc + b) ----------------
__global__ __launch_bounds__(256)
void k_finalize(const float* __restrict__ acc, const float* __restrict__ dinv,
                const float* __restrict__ bias, float* __restrict__ out,
                int n, int do_relu)
{
    int idx = blockIdx.x * 256 + threadIdx.x;      // over n*32 float4s
    if (idx >= n * 32) return;
    int i  = idx >> 5;
    int c4 = idx & 31;
    float di = dinv[i];
    float4 v = ((const float4*)acc)[idx];
    float4 b = ((const float4*)bias)[c4];
    float4 r;
    r.x = fmaf(v.x, di, b.x);
    r.y = fmaf(v.y, di, b.y);
    r.z = fmaf(v.z, di, b.z);
    r.w = fmaf(v.w, di, b.w);
    if (do_relu) {
        r.x = fmaxf(r.x, 0.0f); r.y = fmaxf(r.y, 0.0f);
        r.z = fmaxf(r.z, 0.0f); r.w = fmaxf(r.w, 0.0f);
    }
    ((float4*)out)[idx] = r;
}

// ---------------- launch ----------------
extern "C" void kernel_launch(void* const* d_in, const int* in_sizes, int n_in,
                              void* d_out, int out_size, void* d_ws, size_t ws_size,
                              hipStream_t stream)
{
    const float*     x   = (const float*)d_in[0];
    const long long* ei  = (const long long*)d_in[1];   // int64 (2, E)
    const float*     W1  = (const float*)d_in[2];
    const float*     b1  = (const float*)d_in[3];
    const float*     W2  = (const float*)d_in[4];
    const float*     b2  = (const float*)d_in[5];
    const float*     W3  = (const float*)d_in[6];
    const float*     b3  = (const float*)d_in[7];

    const int N = in_sizes[0] / 256;
    const int E = in_sizes[1] / 2;
    const long long* srcI = ei;        // edge_index[0] = src
    const long long* dstI = ei + E;    // edge_index[1] = dst

    char*  ws   = (char*)d_ws;
    float* dinv = (float*)ws;
    size_t off  = (((size_t)N * 4) + 511) & ~(size_t)511;
    const size_t hbytes = (size_t)N * 128 * 4;           // multiple of 512
    float* bufA = (float*)(ws + off);
    float* bufB = (float*)(ws + off + hbytes);
    unsigned short* wf1 = (unsigned short*)(ws + off + 2 * hbytes);
    unsigned short* wf2 = wf1 + 256 * 128;
    unsigned short* wf3 = wf2 + 128 * 128;
    float* out  = (float*)d_out;

    const int tpb = 256;
    dim3 gN((N + tpb - 1) / tpb);
    dim3 gE((E + tpb - 1) / tpb);
    dim3 gGemm((N + 63) / 64);
    dim3 gScat((E + 7) / 8);
    dim3 gFin((N * 32 + tpb - 1) / tpb);

    // pack weights into WMMA B-fragment order (bf16), L2-resident
    k_pack_w<<<dim3((256 * 128 + tpb - 1) / tpb), tpb, 0, stream>>>(W1, wf1, 256);
    k_pack_w<<<dim3((128 * 128 + tpb - 1) / tpb), tpb, 0, stream>>>(W2, wf2, 128);
    k_pack_w<<<dim3((128 * 128 + tpb - 1) / tpb), tpb, 0, stream>>>(W3, wf3, 128);

    // normalization: dinv = rsqrt(1 + in-degree)
    k_deg_init <<<gN, tpb, 0, stream>>>(dinv, N);
    k_deg_count<<<gE, tpb, 0, stream>>>(dstI, dinv, E);
    k_rsqrt    <<<gN, tpb, 0, stream>>>(dinv, N);

    // layer 1: x(256) -> h1(128), relu
    gemm_wmma<256><<<gGemm, tpb, 0, stream>>>(x, wf1, dinv, bufA, bufB, N);
    k_scatter     <<<gScat, tpb, 0, stream>>>(srcI, dstI, bufA, bufB, E);
    k_finalize    <<<gFin,  tpb, 0, stream>>>(bufB, dinv, b1, bufA, N, 1);

    // layer 2: h1 -> h2, relu   (d_out used as accumulator scratch)
    gemm_wmma<128><<<gGemm, tpb, 0, stream>>>(bufA, wf2, dinv, bufB, out, N);
    k_scatter     <<<gScat, tpb, 0, stream>>>(srcI, dstI, bufB, out, E);
    k_finalize    <<<gFin,  tpb, 0, stream>>>(out, dinv, b2, bufA, N, 1);

    // layer 3: h2 -> out, no relu
    gemm_wmma<128><<<gGemm, tpb, 0, stream>>>(bufA, wf3, dinv, bufB, out, N);
    k_scatter     <<<gScat, tpb, 0, stream>>>(srcI, dstI, bufB, out, E);
    k_finalize    <<<gFin,  tpb, 0, stream>>>(out, dinv, b3, out, N, 0);
}
